// MobileMQA_71597104824785
// MI455X (gfx1250) — compile-verified, matrix operand order checked by reading, not run on hardware
//
#include <hip/hip_runtime.h>
#include <hip/hip_bf16.h>
#include <stdint.h>

// ---------------- problem constants ----------------
#define B_   8
#define C_   256
#define HH_  48
#define WW_  48
#define N_   (HH_ * WW_)      // 2304 = 144 * 16
#define HD_  256              // NUM_HEADS * HEAD_DIM
#define NH_  4
#define D_   64
#define EPS_ 1e-5f

typedef __attribute__((ext_vector_type(16))) __bf16    v16bf;
typedef __attribute__((ext_vector_type(8)))  float     v8f;
typedef __attribute__((ext_vector_type(4)))  uint32_t  u32x4;
typedef __attribute__((ext_vector_type(8)))  uint32_t  u32x8;

// ---------------------------------------------------------------------------
// Fragment loader for 16-bit A/B operands of v_wmma_*_16x16x32_bf16.
// Memory layout assumed: 16 rows x K, row-major, contiguous along K.
// Per ISA 7.12.2 (16-bit A 16x32):
//   lanes 0-15 : row = lane,    VGPR0-3 = K {0..7} pairs,  VGPR4-7 = K {16..23}
//   lanes 16-31: row = lane-16, VGPR0-3 = K {8..15},       VGPR4-7 = K {24..31}
// => per lane: two contiguous 16-byte chunks -> two b128 loads (global or ds).
// ---------------------------------------------------------------------------
static __device__ __forceinline__ v16bf load_frag_b16(const __bf16* base, int ld,
                                                      int k0, int lane) {
  const int r  = lane & 15;
  const int hi = lane >> 4;
  const uint4* p = reinterpret_cast<const uint4*>(base + (size_t)r * ld + k0);
  union { v16bf v; uint4 q[2]; } f;
  f.q[0] = p[hi];       // K = hi*8 + 0..7
  f.q[1] = p[2 + hi];   // K = 16 + hi*8 + 0..7
  return f.v;
}

static __device__ __forceinline__ v8f wmma_bf16(v16bf a, v16bf b, v8f c) {
  return __builtin_amdgcn_wmma_f32_16x16x32_bf16(false, a, false, b,
                                                 (short)0, c, false, false);
}

// CDNA5 async global->LDS copy (ASYNCcnt-tracked), GVS addressing mode:
// LDS address comes from vdst; global address is saddr + vaddr
static __device__ __forceinline__ void async_b128(void* lds_ptr,
                                                  const void* gbase,
                                                  uint32_t byte_off) {
  uint32_t lds_off = (uint32_t)(uintptr_t)lds_ptr;  // LDS aperture: offset in [31:0]
  asm volatile("global_load_async_to_lds_b128 %0, %1, %2"
               :: "v"(lds_off), "v"(byte_off), "s"(gbase)
               : "memory");
}

// ---------------------------------------------------------------------------
// CDNA5 Tensor Data Mover: DMA a 2-D tile (rows x cols, 2-byte elements,
// row stride == cols) from global memory into LDS. D# per ISA 08 §8.3/8.4,
// groups 2/3 omitted (<=2-D tensor). TENSORcnt-tracked.
// ---------------------------------------------------------------------------
static __device__ __forceinline__ void tdm_load_2d_b16(void* lds_ptr,
                                                       const void* gptr,
                                                       uint32_t rows,
                                                       uint32_t cols) {
  uint64_t ga  = (uint64_t)(uintptr_t)gptr;
  uint32_t lds = (uint32_t)(uintptr_t)lds_ptr;
  u32x4 g0;
  g0[0] = 1u;                                               // count=1 (valid), user mode
  g0[1] = lds;                                              // lds_addr (bytes)
  g0[2] = (uint32_t)ga;                                     // global_addr[31:0]
  g0[3] = (uint32_t)((ga >> 32) & 0x1FFFFFFu) | (2u << 30); // addr[56:32] | type=2
  u32x8 g1;
  g1[0] = 1u << 16;                       // data_size=1 (2 bytes); no multicast
  g1[1] = (cols & 0xFFFFu) << 16;         // tensor_dim0[15:0]
  g1[2] = (cols >> 16) | ((rows & 0xFFFFu) << 16);  // dim0[31:16] | tensor_dim1[15:0]
  g1[3] = (rows >> 16) | ((cols & 0xFFFFu) << 16);  // dim1[31:16] | tile_dim0
  g1[4] = rows & 0xFFFFu;                 // tile_dim1 (tile_dim2 = 0)
  g1[5] = cols;                           // tensor_dim0_stride[31:0]
  g1[6] = 0u;                             // stride[47:32] | tensor_dim1_stride lo
  g1[7] = 0u;
  asm volatile("tensor_load_to_lds %0, %1" :: "s"(g0), "s"(g1) : "memory");
}

// ---------------- conversion kernels ----------------
__global__ void cvt_f32_bf16(const float* __restrict__ in,
                             __bf16* __restrict__ out, int n) {
  int i = blockIdx.x * blockDim.x + threadIdx.x;
  if (i < n) out[i] = (__bf16)in[i];
}

// x [B][C][N] f32  ->  xt [B][N][C] bf16
__global__ void cvt_x_transpose(const float* __restrict__ x,
                                __bf16* __restrict__ xt) {
  int idx = blockIdx.x * blockDim.x + threadIdx.x;
  if (idx >= B_ * C_ * N_) return;
  int n = idx % N_;
  int c = (idx / N_) % C_;
  int b = idx / (C_ * N_);
  xt[((size_t)b * N_ + n) * C_ + c] = (__bf16)x[idx];
}

// ---------------- fused QKV projection ----------------
// Y[384][N] = wqkv[384][256] * xf[256][N] per batch (rows: 0-255 Q, 256-319 K, 320-383 V)
// Block = 4 waves sharing one 16x256 weight strip, staged once via TDM into LDS;
// each wave computes a 16x64 strip (4 N-tiles), A-fragments come from LDS.
__global__ void gemm_qkv_wmma(const __bf16* __restrict__ wqkv,
                              const __bf16* __restrict__ xt,
                              __bf16* __restrict__ qT,
                              __bf16* __restrict__ kT,
                              __bf16* __restrict__ v) {
  __shared__ __align__(16) __bf16 As[16][C_];  // 8 KB weight strip
  const int lane = threadIdx.x;
  const int wj   = threadIdx.y;                // wave -> n-strip within block
  const int MT = 384 / 16, NGB = (N_ / 64) / 4;  // 24 x 9 block grid (exact)
  int gid = blockIdx.x;
  int b   = gid / (MT * NGB);
  int rm  = gid % (MT * NGB);
  int mt  = rm / NGB, ngb = rm % NGB;
  int ng  = ngb * 4 + wj;

  if (wj == 0) {  // one TDM descriptor per block stages the weight strip
    tdm_load_2d_b16(&As[0][0], wqkv + (size_t)mt * 16 * C_, 16u, (uint32_t)C_);
    asm volatile("s_wait_tensorcnt 0x0" ::: "memory");
  }
  __syncthreads();

  const __bf16* Bb = xt + ((size_t)b * N_ + ng * 64) * C_;

  const v8f vzero = {0.f, 0.f, 0.f, 0.f, 0.f, 0.f, 0.f, 0.f};
  v8f acc[4];
#pragma unroll
  for (int j = 0; j < 4; ++j) acc[j] = vzero;

#pragma unroll
  for (int k0 = 0; k0 < C_; k0 += 32) {
    v16bf a = load_frag_b16(&As[0][0], C_, k0, lane);
    if (k0 + 32 < C_)
      __builtin_prefetch(Bb + k0 + 32, 0, 1);   // global_prefetch_b8
#pragma unroll
    for (int j = 0; j < 4; ++j) {
      v16bf bb = load_frag_b16(Bb + (size_t)j * 16 * C_, C_, k0, lane);
      acc[j] = wmma_bf16(a, bb, acc[j]);
    }
  }

  const int o0 = mt * 16;
  const int col = lane & 15, hi = lane >> 4;
#pragma unroll
  for (int j = 0; j < 4; ++j) {
    int n = ng * 64 + j * 16 + col;
#pragma unroll
    for (int r = 0; r < 8; ++r) {
      int o = o0 + r + hi * 8;
      __bf16 val = (__bf16)acc[j][r];
      if (o0 < 256) {
        qT[((size_t)b * N_ + n) * HD_ + o] = val;
      } else if (o0 < 320) {
        kT[((size_t)b * N_ + n) * D_ + (o - 256)] = val;
      } else {
        v[(size_t)b * D_ * N_ + (size_t)(o - 320) * N_ + n] = val;
      }
    }
  }
}

// ---------------- flash attention (MQA, shared KV head) ----------------
// Block = 4 waves = 4 heads of one (batch, 16-query tile). K/V tiles are shared
// by all heads, staged global->LDS with CDNA5 async copies, double-buffered so
// the async DMA of tile i+1 overlaps WMMA+softmax on tile i.
__global__ void attn_flash_wmma(const __bf16* __restrict__ qT,
                                const __bf16* __restrict__ kT,
                                const __bf16* __restrict__ v,
                                __bf16* __restrict__ aoT) {
  __shared__ __align__(16) __bf16 Kt[2][32][64];  // 8 KB  (32 keys x d=64)
  __shared__ __align__(16) __bf16 Vt[2][64][32];  // 8 KB  (d=64  x 32 keys)
  __shared__ __align__(16) __bf16 Pt[4][16][32];  // 4 KB  per-wave P tiles

  const int lane = threadIdx.x;
  const int h    = threadIdx.y;           // head = wave id
  const int tid  = h * 32 + lane;         // 0..127
  const int QT = N_ / 16;
  int gid = blockIdx.x;                   // exactly B_*QT blocks
  int b = gid / QT, qt = gid % QT;

  const __bf16* qb = qT + ((size_t)b * N_ + qt * 16) * HD_ + h * D_;
  const __bf16* kb = kT + (size_t)b * N_ * D_;
  const __bf16* vb = v + (size_t)b * D_ * N_;
  __bf16* pp = &Pt[h][0][0];

  // cooperative async stage of one 32-key K/V tile (4 b128 per thread)
  auto issue_tile = [&](int buf, int m0) {
#pragma unroll
    for (int j = 0; j < 2; ++j) {         // K tile: 256 chunks of 16B
      int ch = tid * 2 + j;
      int row = ch >> 3, c8 = ch & 7;
      async_b128(&Kt[buf][row][c8 * 8], kb,
                 (uint32_t)(((m0 + row) * D_ + c8 * 8) * 2));
    }
#pragma unroll
    for (int j = 0; j < 2; ++j) {         // V tile: 256 chunks of 16B
      int ch = tid * 2 + j;
      int row = ch >> 2, c8 = ch & 3;
      async_b128(&Vt[buf][row][c8 * 8], vb,
                 (uint32_t)((row * N_ + m0 + c8 * 8) * 2));
    }
  };

  // query A-fragments are loop-invariant
  v16bf qa0 = load_frag_b16(qb, HD_, 0, lane);
  v16bf qa1 = load_frag_b16(qb, HD_, 32, lane);

  const float scale = 0.125f;  // 64^-0.5
  float mrow[8], lrow[8];
  v8f acc[4];
  const v8f vzero = {0.f, 0.f, 0.f, 0.f, 0.f, 0.f, 0.f, 0.f};
#pragma unroll
  for (int r = 0; r < 8; ++r) { mrow[r] = -1e30f; lrow[r] = 0.f; }
#pragma unroll
  for (int t = 0; t < 4; ++t) acc[t] = vzero;

  const int col = lane & 15, hi = lane >> 4;
  const int NTILES = N_ / 32;  // 72

  issue_tile(0, 0);
  for (int it = 0; it < NTILES; ++it) {
    const int cur = it & 1;
    if (it + 1 < NTILES) {
      issue_tile(1 - cur, (it + 1) * 32);
      asm volatile("s_wait_asynccnt 0x4" ::: "memory");  // tile `it` done, next in flight
    } else {
      asm volatile("s_wait_asynccnt 0x0" ::: "memory");
    }
    __syncthreads();  // tile visible to all 4 heads

    // ---- scores: two 16x16 tiles vs LDS K, K-dim = 64 ----
    v8f s0 = vzero, s1 = vzero;
    {
      v16bf b0 = load_frag_b16(&Kt[cur][0][0], D_, 0, lane);
      v16bf b1 = load_frag_b16(&Kt[cur][0][0], D_, 32, lane);
      s0 = wmma_bf16(qa0, b0, s0);
      s0 = wmma_bf16(qa1, b1, s0);
      v16bf c0 = load_frag_b16(&Kt[cur][16][0], D_, 0, lane);
      v16bf c1 = load_frag_b16(&Kt[cur][16][0], D_, 32, lane);
      s1 = wmma_bf16(qa0, c0, s1);
      s1 = wmma_bf16(qa1, c1, s1);
    }

    // ---- online softmax (row = r + hi*8 lives across a 16-lane group) ----
#pragma unroll
    for (int r = 0; r < 8; ++r) {
      float x0 = s0[r] * scale, x1 = s1[r] * scale;
      float mx = fmaxf(x0, x1);
      mx = fmaxf(mx, __shfl_xor(mx, 1, 32));
      mx = fmaxf(mx, __shfl_xor(mx, 2, 32));
      mx = fmaxf(mx, __shfl_xor(mx, 4, 32));
      mx = fmaxf(mx, __shfl_xor(mx, 8, 32));
      float mnew = fmaxf(mrow[r], mx);
      float corr = __expf(mrow[r] - mnew);
      float p0 = __expf(x0 - mnew);
      float p1 = __expf(x1 - mnew);
      float rs = p0 + p1;
      rs += __shfl_xor(rs, 1, 32);
      rs += __shfl_xor(rs, 2, 32);
      rs += __shfl_xor(rs, 4, 32);
      rs += __shfl_xor(rs, 8, 32);
      lrow[r] = lrow[r] * corr + rs;
      mrow[r] = mnew;
#pragma unroll
      for (int t = 0; t < 4; ++t) acc[t][r] *= corr;
      int row = r + hi * 8;
      pp[row * 32 + col]      = (__bf16)p0;  // C-layout -> A-layout via LDS
      pp[row * 32 + 16 + col] = (__bf16)p1;
    }
    asm volatile("s_wait_dscnt 0x0" ::: "memory");

    // ---- P @ V: A = P[16x32] (LDS), B = Vt[d-tile][32] (LDS) ----
    v16bf pa = load_frag_b16(pp, 32, 0, lane);
#pragma unroll
    for (int t = 0; t < 4; ++t) {
      v16bf bv = load_frag_b16(&Vt[cur][t * 16][0], 32, 0, lane);
      acc[t] = wmma_bf16(pa, bv, acc[t]);
    }
    __syncthreads();  // everyone done reading before buffer is overwritten
  }

  // ---- finalize: divide by row sums, write aoT[b][n][h*64+d] (bf16) ----
#pragma unroll
  for (int r = 0; r < 8; ++r) {
    float inv = 1.0f / (lrow[r] + 1e-20f);
#pragma unroll
    for (int t = 0; t < 4; ++t) acc[t][r] *= inv;
  }
#pragma unroll
  for (int t = 0; t < 4; ++t) {
#pragma unroll
    for (int r = 0; r < 8; ++r) {
      int n = qt * 16 + r + hi * 8;
      int o = h * D_ + t * 16 + col;
      aoT[((size_t)b * N_ + n) * HD_ + o] = (__bf16)acc[t][r];
    }
  }
}

// ---------------- output projection + bias -> f32 pre-BN buffer ----------------
// Same TDM-staged weight-strip scheme as the QKV GEMM.
__global__ void gemm_proj_wmma(const __bf16* __restrict__ wp,
                               const __bf16* __restrict__ aoT,
                               const float* __restrict__ pb,
                               float* __restrict__ pre) {
  __shared__ __align__(16) __bf16 As[16][HD_];  // 8 KB weight strip
  const int lane = threadIdx.x;
  const int wj   = threadIdx.y;
  const int MT = C_ / 16, NGB = (N_ / 64) / 4;  // 16 x 9 block grid (exact)
  int gid = blockIdx.x;
  int b   = gid / (MT * NGB);
  int rm  = gid % (MT * NGB);
  int mt  = rm / NGB, ngb = rm % NGB;
  int ng  = ngb * 4 + wj;

  if (wj == 0) {
    tdm_load_2d_b16(&As[0][0], wp + (size_t)mt * 16 * HD_, 16u, (uint32_t)HD_);
    asm volatile("s_wait_tensorcnt 0x0" ::: "memory");
  }
  __syncthreads();

  const __bf16* Bb = aoT + ((size_t)b * N_ + ng * 64) * HD_;

  const v8f vzero = {0.f, 0.f, 0.f, 0.f, 0.f, 0.f, 0.f, 0.f};
  v8f acc[4];
#pragma unroll
  for (int j = 0; j < 4; ++j) acc[j] = vzero;

#pragma unroll
  for (int k0 = 0; k0 < HD_; k0 += 32) {
    v16bf a = load_frag_b16(&As[0][0], HD_, k0, lane);
    if (k0 + 32 < HD_)
      __builtin_prefetch(Bb + k0 + 32, 0, 1);
#pragma unroll
    for (int j = 0; j < 4; ++j) {
      v16bf bb = load_frag_b16(Bb + (size_t)j * 16 * HD_, HD_, k0, lane);
      acc[j] = wmma_bf16(a, bb, acc[j]);
    }
  }

  const int col = lane & 15, hi = lane >> 4;
#pragma unroll
  for (int j = 0; j < 4; ++j) {
    int n = ng * 64 + j * 16 + col;
#pragma unroll
    for (int r = 0; r < 8; ++r) {
      int c = mt * 16 + r + hi * 8;
      pre[(size_t)b * C_ * N_ + (size_t)c * N_ + n] = acc[j][r] + pb[c];
    }
  }
}

// ---------------- batch-norm statistics (one block per channel) ----------------
__global__ void bn_stats(const float* __restrict__ pre,
                         float* __restrict__ mean, float* __restrict__ invstd) {
  const int c = blockIdx.x;
  const int tid = threadIdx.x;
  const int TOT = B_ * N_;
  float s = 0.f, ss = 0.f;
  for (int i = tid; i < TOT; i += blockDim.x) {
    int b = i / N_, n = i % N_;
    float vv = pre[(size_t)b * C_ * N_ + (size_t)c * N_ + n];
    s += vv;
    ss += vv * vv;
  }
  __shared__ float sh[256], sh2[256];
  sh[tid] = s; sh2[tid] = ss;
  __syncthreads();
  for (int st = 128; st > 0; st >>= 1) {
    if (tid < st) { sh[tid] += sh[tid + st]; sh2[tid] += sh2[tid + st]; }
    __syncthreads();
  }
  if (tid == 0) {
    float m = sh[0] / (float)TOT;
    float var = sh2[0] / (float)TOT - m * m;  // biased, matches jnp.var
    mean[c] = m;
    invstd[c] = rsqrtf(var + EPS_);
  }
}

// ---------------- BN apply + residual ----------------
__global__ void bn_apply(const float* __restrict__ pre, const float* __restrict__ x,
                         const float* __restrict__ gamma, const float* __restrict__ beta,
                         const float* __restrict__ mean, const float* __restrict__ invstd,
                         float* __restrict__ out) {
  int idx = blockIdx.x * blockDim.x + threadIdx.x;
  if (idx >= B_ * C_ * N_) return;
  int c = (idx / N_) % C_;
  out[idx] = gamma[c] * (pre[idx] - mean[c]) * invstd[c] + beta[c] + x[idx];
}

// ---------------- host launcher ----------------
extern "C" void kernel_launch(void* const* d_in, const int* in_sizes, int n_in,
                              void* d_out, int out_size, void* d_ws, size_t ws_size,
                              hipStream_t stream) {
  (void)in_sizes; (void)n_in; (void)out_size; (void)ws_size;
  const float* x     = (const float*)d_in[0];
  const float* qw    = (const float*)d_in[1];
  const float* kw    = (const float*)d_in[2];
  const float* vw    = (const float*)d_in[3];
  const float* pw    = (const float*)d_in[4];
  const float* pb    = (const float*)d_in[5];
  const float* gamma = (const float*)d_in[6];
  const float* beta  = (const float*)d_in[7];
  float* out = (float*)d_out;

  char* ws = (char*)d_ws;
  size_t off = 0;
  auto alloc = [&](size_t bytes) -> void* {
    void* p = (void*)(ws + off);
    off += (bytes + 255) & ~(size_t)255;
    return p;
  };
  __bf16* xt     = (__bf16*)alloc((size_t)B_ * N_ * C_ * 2);
  __bf16* wqkv   = (__bf16*)alloc((size_t)384 * C_ * 2);
  __bf16* wph    = (__bf16*)alloc((size_t)C_ * HD_ * 2);
  __bf16* qT     = (__bf16*)alloc((size_t)B_ * N_ * HD_ * 2);
  __bf16* kT     = (__bf16*)alloc((size_t)B_ * N_ * D_ * 2);
  __bf16* vbuf   = (__bf16*)alloc((size_t)B_ * D_ * N_ * 2);
  __bf16* aoT    = (__bf16*)alloc((size_t)B_ * N_ * HD_ * 2);
  float*  pre    = (float*)alloc((size_t)B_ * C_ * N_ * 4);
  float*  mean   = (float*)alloc((size_t)C_ * 4);
  float*  invstd = (float*)alloc((size_t)C_ * 4);

  const int total = B_ * C_ * N_;
  cvt_x_transpose<<<(total + 255) / 256, 256, 0, stream>>>(x, xt);
  cvt_f32_bf16<<<(256 * 256 + 255) / 256, 256, 0, stream>>>(qw, wqkv, 256 * 256);
  cvt_f32_bf16<<<(64 * 256 + 255) / 256, 256, 0, stream>>>(kw, wqkv + 256 * 256, 64 * 256);
  cvt_f32_bf16<<<(64 * 256 + 255) / 256, 256, 0, stream>>>(vw, wqkv + 320 * 256, 64 * 256);
  cvt_f32_bf16<<<(256 * 256 + 255) / 256, 256, 0, stream>>>(pw, wph, 256 * 256);

  dim3 wblk(32, 4);  // 4 waves per workgroup

  // block = (b, weight strip mt, group of 4 n-strips); exact grid, no early exit
  gemm_qkv_wmma<<<B_ * (384 / 16) * ((N_ / 64) / 4), wblk, 0, stream>>>(wqkv, xt, qT, kT, vbuf);

  // one block per (batch, 16-query tile); 4 waves = 4 heads sharing K/V in LDS
  attn_flash_wmma<<<B_ * (N_ / 16), wblk, 0, stream>>>(qT, kT, vbuf, aoT);

  gemm_proj_wmma<<<B_ * (C_ / 16) * ((N_ / 64) / 4), wblk, 0, stream>>>(wph, aoT, pb, pre);

  bn_stats<<<C_, 256, 0, stream>>>(pre, mean, invstd);
  bn_apply<<<(total + 255) / 256, 256, 0, stream>>>(pre, x, gamma, beta, mean, invstd, out);
}
